// LeViT_89309549953417
// MI455X (gfx1250) — matrix operand order repaired and checked by reading, not compiled, not run on hardware
//
#include <hip/hip_runtime.h>
#include <hip/hip_bf16.h>
#include <cstdint>
#include <cstddef>

typedef _Float16 h16;
typedef __attribute__((ext_vector_type(16))) _Float16 v16h;
typedef __attribute__((ext_vector_type(8)))  float    v8f;
typedef unsigned int v4u  __attribute__((ext_vector_type(4)));
typedef unsigned int u32x4 __attribute__((ext_vector_type(4)));
typedef int          i32x4 __attribute__((ext_vector_type(4)));
typedef int          i32x8 __attribute__((ext_vector_type(8)));

#if defined(__has_builtin) && __has_builtin(__builtin_amdgcn_tensor_load_to_lds)
#define HAVE_TDM 1
#else
#define HAVE_TDM 0
#endif

union FragH { v16h v; v4u q[2]; };

__device__ __forceinline__ float hswish(float x) {
    float r = fminf(fmaxf(x + 3.0f, 0.0f), 6.0f);
    return x * r * 0.16666667f;
}

// A fragment (16-bit A 16x32 layout): comps 0..7 = [k+hs*8 .. +7], 8..15 = [k+16+hs*8 .. +7]
__device__ __forceinline__ v16h fragA(const h16* p) {
    const v4u* q = (const v4u*)p;
    FragH u; u.q[0] = q[0]; u.q[1] = q[2];
    return u.v;
}
// B fragment from a row-major [N,K] strip: comps 0..7 = [n, k+hs*8 .. +7], 8..15 = [n, k+16+hs*8 .. +7]
__device__ __forceinline__ v16h fragB(const h16* p) {
    const v4u* q = (const v4u*)p;
    FragH u; u.q[0] = q[0]; u.q[1] = q[2];
    return u.v;
}

// -------------------- generic batched WMMA GEMM with fused epilogue -------------
// C[z] = act( (A[z][M,K] @ Bt[z][N,K]^T) * scale[n] + bias[n] ) (+ R)
// A row-major [M,K] f16 (global), Bt row-major [N,K] f16, K % 32 == 0, K <= 512.
// batch offset for X in {A,Bt,C}: (z/inner)*sXo + (z%inner)*sXi  (element units)
// blockDim=128: the block's 64-row Bt strip is staged into LDS once via the
// Tensor Data Mover (tensor_load_to_lds + s_wait_tensorcnt), then 4 waves x
// 4 M-subtiles read B fragments from LDS (ds_load_b128) while A streams from
// global. Out-of-range rows/cols use CLAMPED indices; garbage results are
// discarded at store time so EXEC stays full for every v_wmma.
__global__ void wmma_gemm(
    const h16* __restrict__ A,  int lda, size_t sAo, size_t sAi,
    const h16* __restrict__ Bt, int ldb, size_t sBo, size_t sBi,
    float* __restrict__ Cf, h16* __restrict__ Ch, int ldc, size_t sCo, size_t sCi,
    const float* __restrict__ R,
    const float* __restrict__ scale, const float* __restrict__ bias,
    int M, int N, int K, int inner, int act)
{
    __shared__ h16 bsh[64 * 512];                 // 64KB: block's Bt strip [64 x K]

    const int lane   = threadIdx.x & 31;
    const int wav    = threadIdx.x >> 5;
    const int nblock = blockIdx.x * 64;           // block's N strip start (< N always)
    const int nbase  = nblock + wav * 16;
    const int mbase  = blockIdx.y * 64;           // 4 M-subtiles per wave
    const int z  = blockIdx.z;
    const int zo = z / inner, zi = z % inner;

    const h16* Ab = A  + (size_t)zo * sAo + (size_t)zi * sAi;
    const h16* Bb = Bt + (size_t)zo * sBo + (size_t)zi * sBi;

    // ---- stage Bt rows [nblock, nblock+64) x [0,K) into LDS ----
#if HAVE_TDM
    if (wav == 0) {
        const unsigned long long ga =
            (unsigned long long)(uintptr_t)(Bb + (size_t)nblock * ldb);
        const unsigned ldsa = (unsigned)(uintptr_t)&bsh[0];   // LDS byte offset
        // D# group0: count=1 | lds_addr | global_addr[56:0] | type=2
        u32x4 g0 = { 1u, ldsa, (unsigned)ga,
                     (unsigned)((ga >> 32) & 0x01FFFFFFu) | (2u << 30) };
        // D# group1: data_size=2B; tensor_dim0=K, tensor_dim1=64;
        // tile_dim0=K, tile_dim1=64; tensor_dim0_stride=ldb (element units)
        i32x8 g1 = { (int)(1u << 16),
                     (int)(((unsigned)K & 0xFFFFu) << 16),
                     (int)(((unsigned)K >> 16) | (64u << 16)),
                     (int)(((unsigned)K & 0xFFFFu) << 16),
                     (int)64,
                     (int)(unsigned)ldb,
                     0, 0 };
        i32x4 gz = { 0, 0, 0, 0 };
#if __clang_major__ >= 23
        i32x8 gz8 = { 0, 0, 0, 0, 0, 0, 0, 0 };
        __builtin_amdgcn_tensor_load_to_lds(g0, g1, gz, gz, gz8, 0);
#else
        __builtin_amdgcn_tensor_load_to_lds(g0, g1, gz, gz, 0);
#endif
        __builtin_amdgcn_s_wait_tensorcnt(0);
    }
#else
    {   // fallback: cooperative 16B-chunk copy (global -> LDS)
        const int tid = threadIdx.x;
        const int cpr = K >> 3;                   // 16B chunks per row
        for (int idx = tid; idx < 64 * cpr; idx += 128) {
            int row = idx / cpr, ch = idx - row * cpr;
            int gr = min(nblock + row, N - 1);
            *(v4u*)(bsh + (size_t)row * K + (ch << 3)) =
                *(const v4u*)(Bb + (size_t)gr * ldb + (ch << 3));
        }
    }
#endif
    __syncthreads();

    const int hs  = lane >> 4;
    const int l15 = lane & 15;

    // clamped per-lane pointers (always legal addresses)
    const h16* Ar0 = Ab + (size_t)min(mbase +  0 + l15, M - 1) * lda + hs * 8;
    const h16* Ar1 = Ab + (size_t)min(mbase + 16 + l15, M - 1) * lda + hs * 8;
    const h16* Ar2 = Ab + (size_t)min(mbase + 32 + l15, M - 1) * lda + hs * 8;
    const h16* Ar3 = Ab + (size_t)min(mbase + 48 + l15, M - 1) * lda + hs * 8;
    const int  brow = min(nbase + l15, N - 1) - nblock;       // 0..63
    const h16* Brl  = bsh + (size_t)brow * K + hs * 8;        // LDS

    v8f acc0 = {}, acc1 = {}, acc2 = {}, acc3 = {};
    for (int k0 = 0; k0 < K; k0 += 32) {
        v16h bv  = fragB(Brl + k0);               // ds_load_b128 x2
        v16h a0v = fragA(Ar0 + k0);
        v16h a1v = fragA(Ar1 + k0);
        v16h a2v = fragA(Ar2 + k0);
        v16h a3v = fragA(Ar3 + k0);
        if (k0 + 32 < K) {
            __builtin_prefetch(Ar0 + k0 + 32, 0, 3);
            __builtin_prefetch(Ar2 + k0 + 32, 0, 3);
        }
        acc0 = __builtin_amdgcn_wmma_f32_16x16x32_f16(false, a0v, false, bv, (short)0, acc0, false, false);
        acc1 = __builtin_amdgcn_wmma_f32_16x16x32_f16(false, a1v, false, bv, (short)0, acc1, false, false);
        acc2 = __builtin_amdgcn_wmma_f32_16x16x32_f16(false, a2v, false, bv, (short)0, acc2, false, false);
        acc3 = __builtin_amdgcn_wmma_f32_16x16x32_f16(false, a3v, false, bv, (short)0, acc3, false, false);
    }

    const int cn = nbase + l15;                   // lane's C column
    if (cn >= N) return;                          // after barrier: divergence OK
    const float sc = scale ? scale[cn] : 1.0f;
    const float bi = bias  ? bias[cn]  : 0.0f;
    const size_t cb = (size_t)zo * sCo + (size_t)zi * sCi;
    v8f accs[4] = {acc0, acc1, acc2, acc3};
#pragma unroll
    for (int i = 0; i < 4; ++i) {
#pragma unroll
        for (int r = 0; r < 8; ++r) {
            int cm = mbase + i * 16 + hs * 8 + r; // lane's C row for VGPR r
            if (cm < M) {
                float v = accs[i][r] * sc + bi;
                if (act) v = hswish(v);
                size_t o = cb + (size_t)cm * ldc + cn;
                if (R)  v += R[o];
                if (Cf) Cf[o] = v;
                if (Ch) Ch[o] = (h16)v;
            }
        }
    }
}

// -------------------- stem: 3x3 stride-2 SAME conv (pad lo=0, hi=1) ------------
__global__ void conv3x3s2(const float* __restrict__ in, const float* __restrict__ w,
                          const float* __restrict__ bias, float* __restrict__ out,
                          int Bn, int H, int W, int Ci, int Co)
{
    int Ho = H >> 1, Wo = W >> 1;
    size_t idx = (size_t)blockIdx.x * blockDim.x + threadIdx.x;
    size_t total = (size_t)Bn * Ho * Wo * Co;
    if (idx >= total) return;
    int co = (int)(idx % Co); size_t t = idx / Co;
    int ox = (int)(t % Wo); t /= Wo;
    int oy = (int)(t % Ho); int b = (int)(t / Ho);
    float acc = bias[co];
    for (int ky = 0; ky < 3; ++ky) {
        int iy = 2 * oy + ky;
        if (iy >= H) continue;
        for (int kx = 0; kx < 3; ++kx) {
            int ix = 2 * ox + kx;
            if (ix >= W) continue;
            const float* ip = in + (((size_t)b * H + iy) * W + ix) * Ci;
            const float* wp = w + (size_t)(ky * 3 + kx) * Ci * Co + co;
            for (int ci = 0; ci < Ci; ++ci)
                acc = fmaf(ip[ci], wp[(size_t)ci * Co], acc);
        }
    }
    out[idx] = acc;
}

// -------------------- small helpers --------------------------------------------
__global__ void cvt_f32_f16(const float* __restrict__ a, h16* __restrict__ o, size_t n) {
    size_t i = (size_t)blockIdx.x * blockDim.x + threadIdx.x;
    if (i < n) o[i] = (h16)a[i];
}

// W[K,N] f32 row-major -> out[N,K] f16 (transposed weight for WMMA B operand)
__global__ void convT_f32_f16(const float* __restrict__ w, h16* __restrict__ o, int K, int N) {
    size_t idx = (size_t)blockIdx.x * blockDim.x + threadIdx.x;
    if (idx >= (size_t)K * N) return;
    int k = (int)(idx % K);
    int n = (int)(idx / K);
    o[idx] = (h16)w[(size_t)k * N + n];
}

__global__ void bn_prep(const float* __restrict__ sc, const float* __restrict__ mean,
                        const float* __restrict__ var, float* __restrict__ s,
                        float* __restrict__ b, int n) {
    int i = blockIdx.x * blockDim.x + threadIdx.x;
    if (i < n) {
        float t = sc[i] * rsqrtf(var[i] + 1e-5f);
        s[i] = t;
        b[i] = -mean[i] * t;
    }
}

// softmax over 196 cols with additive attention bias; writes f16 rows zero-padded to 224
__global__ void softmax_rows(const float* __restrict__ scores,
                             const float* __restrict__ bias,
                             h16* __restrict__ attnH) {
    const int row = blockIdx.x;       // 0..195
    const int z   = blockIdx.y;       // b*heads + h
    const int t   = threadIdx.x;      // 256 threads
    const float* sr = scores + ((size_t)z * 196 + row) * 196;
    const float* br = bias + (size_t)row * 196;
    __shared__ float red[256];
    float v = (t < 196) ? sr[t] + br[t] : -3.0e38f;
    red[t] = v; __syncthreads();
    for (int s = 128; s > 0; s >>= 1) { if (t < s) red[t] = fmaxf(red[t], red[t + s]); __syncthreads(); }
    float mx = red[0]; __syncthreads();
    float e = (t < 196) ? __expf(v - mx) : 0.0f;
    red[t] = e; __syncthreads();
    for (int s = 128; s > 0; s >>= 1) { if (t < s) red[t] += red[t + s]; __syncthreads(); }
    float inv = 1.0f / red[0];
    h16* orow = attnH + ((size_t)z * 196 + row) * 224;
    if (t < 224) orow[t] = (t < 196) ? (h16)(e * inv) : (h16)0.0f;
}

// vT[z][d][n] = V[b, n, h*64+d]  (f16, n zero-padded to 224) from qkv f16 buffer
__global__ void build_vT(const h16* __restrict__ qkvh, h16* __restrict__ vT,
                         int heads, int hd3, int hd) {
    const int z = blockIdx.y;
    const int b = z / heads, h = z % heads;
    int idx = blockIdx.x * 256 + threadIdx.x;   // 0 .. 64*224-1
    int n = idx % 224;
    int d = idx / 224;
    h16 val = (h16)0.0f;
    if (n < 196)
        val = qkvh[((size_t)b * 196 + n) * hd3 + 2 * hd + h * 64 + d];
    vT[((size_t)z * 64 + d) * 224 + n] = val;
}

// -------------------- orchestration --------------------------------------------
extern "C" void kernel_launch(void* const* d_in, const int* in_sizes, int n_in,
                              void* d_out, int out_size, void* d_ws, size_t ws_size,
                              hipStream_t stream)
{
    (void)in_sizes; (void)n_in; (void)out_size; (void)ws_size;
    const int Bn = 32, TOK = 196, C = 256, FFN = 512;
    const int M = Bn * TOK;                        // 6272 token rows
    const unsigned MY  = (unsigned)(M / 16 / 4);   // 98 blocks of 64 rows
    const unsigned MYA = 4;                        // ceil(13 tiles / 4) for M=196
    static const int HEADS[12] = {4,4,4,4,6,6,6,6,8,8,8,8};

    const float* x_in = (const float*)d_in[0];

    char* ws = (char*)d_ws;
    size_t off = 0;
    auto take = [&](size_t bytes) -> char* {
        char* p = ws + off;
        off += (bytes + 255) & ~(size_t)255;
        return p;
    };
    float* x  = (float*)take((size_t)M * C * 4);   // persistent token state (f32)
    h16*   xh = (h16*)  take((size_t)M * C * 2);   // f16 mirror of x
    char*  S  = ws + off;                          // overlapping scratch region

    // stem scratch (lives only before block phase)
    float* a0 = (float*)S;                                                 // 32*112*112*32
    float* a1 = (float*)(S + (size_t)32*112*112*32*4);                     // 32*56*56*64
    float* a2 = (float*)(S + (size_t)32*112*112*32*4 + (size_t)32*56*56*64*4); // 32*28*28*128

    // block scratch (reuses S)
    size_t boff = 0;
    auto btake = [&](size_t bytes) -> char* {
        char* p = S + boff;
        boff += (bytes + 255) & ~(size_t)255;
        return p;
    };
    h16*   qkvh   = (h16*)  btake((size_t)M * 1536 * 2);
    float* scores = (float*)btake((size_t)256 * TOK * TOK * 4);
    h16*   attnH  = (h16*)  btake((size_t)256 * TOK * 224 * 2);
    h16*   vT     = (h16*)  btake((size_t)256 * 64 * 224 * 2);
    h16*   aoutH  = (h16*)  btake((size_t)M * 512 * 2);
    h16*   ffnH   = (h16*)  btake((size_t)M * FFN * 2);
    h16*   wT     = (h16*)  btake((size_t)1536 * 256 * 2);
    float* bns    = (float*)btake(1536 * 4);
    float* bnb    = (float*)btake(1536 * 4);
    (void)btake(262144);                           // slack for TDM strip overread

    // ---- stem: 4 stride-2 convs, 224 -> 14 ----
    {
        size_t t1 = (size_t)32 * 112 * 112 * 32;
        conv3x3s2<<<(unsigned)((t1 + 255) / 256), 256, 0, stream>>>(
            x_in, (const float*)d_in[1], (const float*)d_in[2], a0, 32, 224, 224, 3, 32);
        size_t t2 = (size_t)32 * 56 * 56 * 64;
        conv3x3s2<<<(unsigned)((t2 + 255) / 256), 256, 0, stream>>>(
            a0, (const float*)d_in[3], (const float*)d_in[4], a1, 32, 112, 112, 32, 64);
        size_t t3 = (size_t)32 * 28 * 28 * 128;
        conv3x3s2<<<(unsigned)((t3 + 255) / 256), 256, 0, stream>>>(
            a1, (const float*)d_in[5], (const float*)d_in[6], a2, 32, 56, 56, 64, 128);
        size_t t4 = (size_t)M * C;
        conv3x3s2<<<(unsigned)((t4 + 255) / 256), 256, 0, stream>>>(
            a2, (const float*)d_in[7], (const float*)d_in[8], x, 32, 28, 28, 128, 256);
        cvt_f32_f16<<<(unsigned)((t4 + 255) / 256), 256, 0, stream>>>(x, xh, t4);
    }

    // ---- 12 transformer blocks ----
    for (int blk = 0; blk < 12; ++blk) {
        const int heads = HEADS[blk], hd = heads * 64, hd3 = 3 * hd;
        const void* const* bp = d_in + 9 + blk * 10;
        const float* qkv_w   = (const float*)bp[0];
        const float* bn_sc   = (const float*)bp[1];
        const float* bn_mean = (const float*)bp[2];
        const float* bn_var  = (const float*)bp[3];
        const float* attn_b  = (const float*)bp[4];
        const float* proj_w  = (const float*)bp[5];
        const float* ffn_w1  = (const float*)bp[6];
        const float* ffn_b1  = (const float*)bp[7];
        const float* ffn_w2  = (const float*)bp[8];
        const float* ffn_b2  = (const float*)bp[9];

        // qkv = BN(x @ qkv_w), stored f16
        bn_prep<<<(hd3 + 255) / 256, 256, 0, stream>>>(bn_sc, bn_mean, bn_var, bns, bnb, hd3);
        convT_f32_f16<<<(unsigned)(((size_t)hd3 * C + 255) / 256), 256, 0, stream>>>(qkv_w, wT, C, hd3);
        wmma_gemm<<<dim3((unsigned)((hd3 / 16 + 3) / 4), MY, 1), 128, 0, stream>>>(
            xh, C, 0, 0, wT, C, 0, 0, nullptr, qkvh, hd3, 0, 0,
            nullptr, bns, bnb, M, hd3, C, 1, 0);

        // scores[z,196,196] = q @ k^T   (z = b*heads + h)
        wmma_gemm<<<dim3(4, MYA, Bn * heads), 128, 0, stream>>>(
            qkvh, hd3, (size_t)TOK * hd3, 64,
            qkvh + hd, hd3, (size_t)TOK * hd3, 64,
            scores, nullptr, TOK, (size_t)heads * TOK * TOK, (size_t)TOK * TOK,
            nullptr, nullptr, nullptr, TOK, TOK, 64, heads, 0);

        // softmax(scores + attn_bias) -> f16 rows padded to K=224
        softmax_rows<<<dim3(TOK, Bn * heads), 256, 0, stream>>>(scores, attn_b, attnH);
        build_vT<<<dim3(56, Bn * heads), 256, 0, stream>>>(qkvh, vT, heads, hd3, hd);

        // out = hard_swish(attn @ v), stored f16 as [B, N, hd]
        wmma_gemm<<<dim3(1, MYA, Bn * heads), 128, 0, stream>>>(
            attnH, 224, (size_t)heads * TOK * 224, (size_t)TOK * 224,
            vT, 224, (size_t)heads * 64 * 224, (size_t)64 * 224,
            nullptr, aoutH, hd, (size_t)TOK * hd, 64,
            nullptr, nullptr, nullptr, TOK, 64, 224, heads, 1);

        // x += out @ proj_w   (dual store f32 + f16)
        convT_f32_f16<<<(unsigned)(((size_t)C * hd + 255) / 256), 256, 0, stream>>>(proj_w, wT, hd, C);
        wmma_gemm<<<dim3(C / 16 / 4, MY, 1), 128, 0, stream>>>(
            aoutH, hd, 0, 0, wT, hd, 0, 0, x, xh, C, 0, 0,
            x, nullptr, nullptr, M, C, hd, 1, 0);

        // h = hard_swish(x @ ffn_w1 + b1), stored f16
        convT_f32_f16<<<(unsigned)(((size_t)FFN * C + 255) / 256), 256, 0, stream>>>(ffn_w1, wT, C, FFN);
        wmma_gemm<<<dim3(FFN / 16 / 4, MY, 1), 128, 0, stream>>>(
            xh, C, 0, 0, wT, C, 0, 0, nullptr, ffnH, FFN, 0, 0,
            nullptr, nullptr, ffn_b1, M, FFN, C, 1, 1);

        // x += h @ ffn_w2 + b2 ; last block writes f32 result directly into d_out
        convT_f32_f16<<<(unsigned)(((size_t)C * FFN + 255) / 256), 256, 0, stream>>>(ffn_w2, wT, FFN, C);
        float* cdst = (blk == 11) ? (float*)d_out : x;
        wmma_gemm<<<dim3(C / 16 / 4, MY, 1), 128, 0, stream>>>(
            ffnH, FFN, 0, 0, wT, FFN, 0, 0, cdst, xh, C, 0, 0,
            x, nullptr, ffn_b2, M, C, FFN, 1, 0);
    }
}
